// EntityMambaLayer_32246614459124
// MI455X (gfx1250) — matrix operand order repaired and checked by reading, not compiled, not run on hardware
//
#include <hip/hip_runtime.h>
#include <hip/hip_bf16.h>

// ---------------------------------------------------------------------------
// EntityMambaLayer for MI455X (gfx1250, wave32, WMMA).
// B=64, S=512, D=512, D_INNER=1024, D_STATE=32, D_CONV=4.
// GEMMs: bf16 v_wmma_f32_16x16x32_bf16, double-buffered LDS fed by
// global_load_async_to_lds_b128 (ASYNCcnt). Scan/LN/conv in fp32 VALU.
// ---------------------------------------------------------------------------

typedef __attribute__((ext_vector_type(16))) __bf16 v16bf;
typedef __attribute__((ext_vector_type(8)))  __bf16 v8bf;
typedef __attribute__((ext_vector_type(8)))  float  v8f;

#define NB 64
#define NS 512
#define ND 512
#define NDI 1024
#define NSTATE 32
#define NROWS (NB * NS)   // 32768

static __device__ inline v16bf mk16(v8bf lo, v8bf hi) {
    v16bf r;
    __builtin_memcpy(&r, &lo, 16);
    __builtin_memcpy((char*)&r + 16, &hi, 16);
    return r;
}

// CDNA5 async global->LDS copy, 16B per lane; tracked by ASYNCcnt.
static __device__ inline void async_ld_b128(unsigned lds_off, const __bf16* gptr) {
    asm volatile("global_load_async_to_lds_b128 %0, %1, off"
                 :: "v"(lds_off), "v"(gptr) : "memory");
}
static __device__ inline void wait_async0() {
    asm volatile("s_wait_asynccnt 0x0" ::: "memory");
}

// -------------------------- weight conversion ------------------------------
__global__ __launch_bounds__(256) void cvt_bf16_kernel(const float* __restrict__ in,
                                                       __bf16* __restrict__ out, int n) {
    int i = blockIdx.x * 256 + threadIdx.x;
    if (i < n) out[i] = (__bf16)in[i];
}

// x_proj_w [65,1024] -> padded [128,1024] bf16 (rows >= 65 zero)
__global__ __launch_bounds__(256) void cvt_pad_w2_kernel(const float* __restrict__ in,
                                                         __bf16* __restrict__ out) {
    int i = blockIdx.x * 256 + threadIdx.x;     // 128*1024 total
    int row = i >> 10, col = i & 1023;
    out[i] = (__bf16)(row < 65 ? in[row * 1024 + col] : 0.0f);
}

// ------------------------------ LayerNorm ----------------------------------
__global__ __launch_bounds__(256) void ln_kernel(const float* __restrict__ x,
                                                 const float* __restrict__ gamma,
                                                 const float* __restrict__ beta,
                                                 __bf16* __restrict__ xn) {
    int row = blockIdx.x;
    int tid = threadIdx.x;
    const float* xr = x + (size_t)row * ND;
    float a0 = xr[tid], a1 = xr[tid + 256];

    __shared__ float red[256];
    red[tid] = a0 + a1;
    __syncthreads();
    for (int s = 128; s > 0; s >>= 1) {
        if (tid < s) red[tid] += red[tid + s];
        __syncthreads();
    }
    float mu = red[0] * (1.0f / ND);
    __syncthreads();
    float d0 = a0 - mu, d1 = a1 - mu;
    red[tid] = d0 * d0 + d1 * d1;
    __syncthreads();
    for (int s = 128; s > 0; s >>= 1) {
        if (tid < s) red[tid] += red[tid + s];
        __syncthreads();
    }
    float rstd = rsqrtf(red[0] * (1.0f / ND) + 1e-5f);
    size_t base = (size_t)row * ND;
    xn[base + tid]       = (__bf16)(d0 * rstd * gamma[tid]       + beta[tid]);
    xn[base + tid + 256] = (__bf16)(d1 * rstd * gamma[tid + 256] + beta[tid + 256]);
}

// ------------------------------ WMMA GEMM ----------------------------------
// C[M,N] = A[M,K] (row-major bf16) @ W[N,K]^T (row-major bf16), fp32 acc.
// Block tile 128x128, 8 waves (2x4), wave tile 64x32 = 4x2 WMMA 16x16 tiles.
// Double-buffered LDS; async global->LDS loads overlap the WMMA work of the
// previous tile; one s_wait_asynccnt + one barrier per K-tile.
// EPI 0: fp32 store (ldc).  EPI 1: split bf16 store (x_in | z, each ld 1024).
// EPI 2: fp32 store + residual add (ldc).
template <int EPI>
__global__ __launch_bounds__(256) void gemm_bf16_wmma(
    const __bf16* __restrict__ A, const __bf16* __restrict__ W, int K,
    float* __restrict__ outF, int ldc,
    __bf16* __restrict__ outA, __bf16* __restrict__ outB, int split,
    const float* __restrict__ res) {
    constexpr int LDT = 40;  // 32 + 8 pad (16B-aligned rows, bank-spread)
    __shared__ __bf16 lsA[2][128 * LDT];
    __shared__ __bf16 lsW[2][128 * LDT];

    int tid = threadIdx.x;
    int lane = tid & 31, wave = tid >> 5;
    int waveM = wave & 1, waveN = wave >> 1;   // 2 x 4 wave grid
    int lh = lane & 15;
    int hiHalf = lane >> 4;                    // 0|1

    size_t mBlock = (size_t)blockIdx.y * 128;
    int nBlock = blockIdx.x * 128;

    // this thread's two staging chunks (8 bf16 = 16B each)
    int chunk0 = tid, chunk1 = tid + 256;      // of 512 chunks per tile
    int r0 = chunk0 >> 2, c0 = (chunk0 & 3) << 3;
    int r1 = chunk1 >> 2, c1 = (chunk1 & 3) << 3;

    auto stage_async = [&](int p, int k0) {
        async_ld_b128((unsigned)(size_t)&lsA[p][r0 * LDT + c0],
                      A + (mBlock + r0) * K + k0 + c0);
        async_ld_b128((unsigned)(size_t)&lsW[p][r0 * LDT + c0],
                      W + (size_t)(nBlock + r0) * K + k0 + c0);
        async_ld_b128((unsigned)(size_t)&lsA[p][r1 * LDT + c1],
                      A + (mBlock + r1) * K + k0 + c1);
        async_ld_b128((unsigned)(size_t)&lsW[p][r1 * LDT + c1],
                      W + (size_t)(nBlock + r1) * K + k0 + c1);
    };

    v8f acc[4][2] = {};

    int nTiles = K >> 5;
    stage_async(0, 0);                         // prologue
    for (int t = 0; t < nTiles; ++t) {
        int cur = t & 1;
        wait_async0();                         // tile t resident (this wave)
        __syncthreads();                       // ... and all waves' pieces
        if (t + 1 < nTiles) stage_async(1 - cur, (t + 1) << 5);  // overlap
        if (t + 2 < nTiles) {                  // far L2 prefetch
            int k2 = (t + 2) << 5;
            __builtin_prefetch(A + (mBlock + r0) * K + k2, 0, 1);
            __builtin_prefetch(W + (size_t)(nBlock + r0) * K + k2, 0, 1);
        }

        // B fragments: lane = column n, K contiguous per lane-half
        v16bf bfr[2];
#pragma unroll
        for (int nt = 0; nt < 2; ++nt) {
            int n = waveN * 32 + nt * 16 + lh;
            const __bf16* p = &lsW[cur][n * LDT + hiHalf * 16];
            bfr[nt] = mk16(*(const v8bf*)p, *(const v8bf*)(p + 8));
        }
#pragma unroll
        for (int mt = 0; mt < 4; ++mt) {
            // A fragment: lanes0-15 K 0-15, lanes16-31 K 16-31
            int m = waveM * 64 + mt * 16 + lh;
            const __bf16* p = &lsA[cur][m * LDT + hiHalf * 16];
            v16bf af = mk16(*(const v8bf*)p, *(const v8bf*)(p + 8));
#pragma unroll
            for (int nt = 0; nt < 2; ++nt) {
                acc[mt][nt] = __builtin_amdgcn_wmma_f32_16x16x32_bf16(
                    false, af, false, bfr[nt], (short)0, acc[mt][nt], false, false);
            }
        }
        // no trailing barrier needed: a wave only reaches the next top
        // barrier after its ds_loads of buf[cur] were consumed by WMMA.
    }

    // epilogue: C/D layout -> VGPR v: row = v + 8*hiHalf, col = lane%16
#pragma unroll
    for (int mt = 0; mt < 4; ++mt)
#pragma unroll
        for (int nt = 0; nt < 2; ++nt)
#pragma unroll
            for (int v = 0; v < 8; ++v) {
                size_t row = mBlock + waveM * 64 + mt * 16 + v + hiHalf * 8;
                int col = nBlock + waveN * 32 + nt * 16 + lh;
                float val = acc[mt][nt][v];
                if (EPI == 0) {
                    outF[row * ldc + col] = val;
                } else if (EPI == 1) {
                    if (col < split) outA[(row << 10) + col] = (__bf16)val;
                    else             outB[(row << 10) + (col - split)] = (__bf16)val;
                } else {
                    outF[row * ldc + col] = val + res[row * ldc + col];
                }
            }
}

// ----------------------- depthwise causal conv + SiLU ----------------------
__global__ __launch_bounds__(256) void conv_silu_kernel(const __bf16* __restrict__ xin,
                                                        const float* __restrict__ cw,
                                                        const float* __restrict__ cb,
                                                        __bf16* __restrict__ xact) {
    size_t idx = (size_t)blockIdx.x * 256 + threadIdx.x;  // B*S*Di
    int d = idx & 1023;
    int s = (int)((idx >> 10) & 511);
    int b = (int)(idx >> 19);
    float acc = cb[d];
#pragma unroll
    for (int w = 0; w < 4; ++w) {
        int sp = s - 3 + w;
        if (sp >= 0)
            acc = fmaf((float)xin[(((size_t)b * NS + sp) << 10) + d], cw[d * 4 + w], acc);
    }
    float sig = 1.0f / (1.0f + __expf(-acc));
    xact[idx] = (__bf16)(acc * sig);
}

// --------------------------- selective scan --------------------------------
// grid (NDI/256, NB); one lane per d_inner channel; h[32] in registers.
// Double-buffered B/C/meta stage: one barrier per sequential step.
__global__ __launch_bounds__(256) void scan_kernel(
    const __bf16* __restrict__ xact, const __bf16* __restrict__ zbuf,
    const float* __restrict__ xssm, const float* __restrict__ ebias,
    const float* __restrict__ dtw, const float* __restrict__ dtb,
    const float* __restrict__ escale, __bf16* __restrict__ yg) {
    int tid = threadIdx.x;
    int d = blockIdx.x * 256 + tid;
    int b = blockIdx.y;
    float w_d = dtw[d], b_d = dtb[d], es = escale[0];

    float h[NSTATE];
#pragma unroll
    for (int n = 0; n < NSTATE; ++n) h[n] = 0.0f;

    __shared__ float sB[2][NSTATE], sC[2][NSTATE], sM[2][2];

    auto stage = [&](int p, int s) {
        size_t rb = (size_t)b * NS + s;
        if (tid < 32)        sB[p][tid]      = xssm[rb * 128 + 1 + tid];
        else if (tid < 64)   sC[p][tid - 32] = xssm[rb * 128 + 33 + (tid - 32)];
        else if (tid == 64)  sM[p][0]        = xssm[rb * 128];
        else if (tid == 65)  sM[p][1]        = ebias[rb];
    };

    stage(0, 0);
    __syncthreads();
    for (int s = 0; s < NS; ++s) {
        int cur = s & 1;
        if (s + 1 < NS) stage(1 - cur, s + 1);   // prefetch next step's B/C

        size_t rb = (size_t)b * NS + s;
        float t = fmaf(sM[cur][0], w_d, b_d);
        float sp = (t > 20.0f) ? t : log1pf(__expf(t));
        float delta = fmaxf(fmaf(es, sM[cur][1], sp), 1e-4f);
        float decay = __expf(-delta);
        float xa = (float)xact[(rb << 10) + d];

        float y = 0.0f;
#pragma unroll
        for (int n = 0; n < NSTATE; ++n) {
            h[n] = fmaf(h[n], decay, sB[cur][n] * xa);
            h[n] = fminf(fmaxf(h[n], -100.0f), 100.0f);
            y = fmaf(h[n], sC[cur][n], y);
        }

        float zf = (float)zbuf[(rb << 10) + d];
        float zg = zf / (1.0f + __expf(-zf));     // silu(z)
        yg[(rb << 10) + d] = (__bf16)(y * zg);
        __syncthreads();
    }
}

// ------------------------------- launcher ----------------------------------
extern "C" void kernel_launch(void* const* d_in, const int* in_sizes, int n_in,
                              void* d_out, int out_size, void* d_ws, size_t ws_size,
                              hipStream_t stream) {
    const float* x     = (const float*)d_in[0];
    const float* eb    = (const float*)d_in[1];
    const float* gamma = (const float*)d_in[2];
    const float* beta  = (const float*)d_in[3];
    const float* w_in  = (const float*)d_in[4];   // [2048, 512]
    const float* cw    = (const float*)d_in[5];   // [1024, 1, 4]
    const float* cb    = (const float*)d_in[6];
    const float* w_x   = (const float*)d_in[7];   // [65, 1024]
    const float* dtw   = (const float*)d_in[8];   // [1024, 1]
    const float* dtb   = (const float*)d_in[9];
    const float* w_out = (const float*)d_in[10];  // [512, 1024]
    const float* esc   = (const float*)d_in[11];
    float* out = (float*)d_out;

    char* ws = (char*)d_ws;
    size_t o = 0;
    auto carve = [&](size_t bytes) {
        void* p = ws + o;
        o += (bytes + 255) & ~(size_t)255;
        return p;
    };
    __bf16* xn_bf  = (__bf16*)carve((size_t)NROWS * ND * 2);        // 32 MB
    __bf16* w1_bf  = (__bf16*)carve((size_t)2048 * 512 * 2);        // 2 MB
    __bf16* w2_bf  = (__bf16*)carve((size_t)128 * 1024 * 2);        // 256 KB (padded)
    __bf16* w3_bf  = (__bf16*)carve((size_t)512 * 1024 * 2);        // 1 MB
    __bf16* xin_bf = (__bf16*)carve((size_t)NROWS * NDI * 2);       // 64 MB
    __bf16* z_bf   = (__bf16*)carve((size_t)NROWS * NDI * 2);       // 64 MB
    __bf16* xact_bf= (__bf16*)carve((size_t)NROWS * NDI * 2);       // 64 MB
    float*  xssm   = (float*) carve((size_t)NROWS * 128 * 4);       // 16 MB
    __bf16* yg_bf  = (__bf16*)carve((size_t)NROWS * NDI * 2);       // 64 MB
    (void)ws_size; (void)in_sizes; (void)n_in; (void)out_size;

    // weight conversions
    cvt_bf16_kernel<<<(2048 * 512 + 255) / 256, 256, 0, stream>>>(w_in, w1_bf, 2048 * 512);
    cvt_pad_w2_kernel<<<(128 * 1024) / 256, 256, 0, stream>>>(w_x, w2_bf);
    cvt_bf16_kernel<<<(512 * 1024 + 255) / 256, 256, 0, stream>>>(w_out, w3_bf, 512 * 1024);

    // LayerNorm -> bf16
    ln_kernel<<<NROWS, 256, 0, stream>>>(x, gamma, beta, xn_bf);

    // in_proj: [32768,512] x [2048,512]^T -> x_in | z (bf16)
    gemm_bf16_wmma<1><<<dim3(2048 / 128, NROWS / 128), 256, 0, stream>>>(
        xn_bf, w1_bf, ND, nullptr, 0, xin_bf, z_bf, 1024, nullptr);

    // depthwise causal conv + SiLU
    conv_silu_kernel<<<(NROWS * NDI) / 256, 256, 0, stream>>>(xin_bf, cw, cb, xact_bf);

    // x_proj: [32768,1024] x [128(pad),1024]^T -> x_ssm fp32 (ld 128)
    gemm_bf16_wmma<0><<<dim3(1, NROWS / 128), 256, 0, stream>>>(
        xact_bf, w2_bf, NDI, xssm, 128, nullptr, nullptr, 0, nullptr);

    // selective scan + SiLU(z) gating -> y_gated bf16
    scan_kernel<<<dim3(NDI / 256, NB), 256, 0, stream>>>(
        xact_bf, z_bf, xssm, eb, dtw, dtb, esc, yg_bf);

    // out_proj + residual: [32768,1024] x [512,1024]^T + x -> out fp32
    gemm_bf16_wmma<2><<<dim3(512 / 128, NROWS / 128), 256, 0, stream>>>(
        yg_bf, w3_bf, NDI, out, ND, nullptr, nullptr, 0, x);
}
